// Expand_36610301231376
// MI455X (gfx1250) — compile-verified, matrix operand order checked by reading, not run on hardware
//
#include <hip/hip_runtime.h>
#include <hip/hip_bf16.h>

// ---------------------------------------------------------------------------
// Problem constants
// ---------------------------------------------------------------------------
#define BATCH   16
#define T_LEN   149
#define K_PAD   160      // T_LEN padded up to 5*32 for WMMA K-loop
#define T_DIM   768
#define HW      56
#define NSP     (HW*HW)  // 3136 spatial tokens
#define S_DIM   512
#define LN_EPS  1e-5f

typedef __attribute__((ext_vector_type(16))) __bf16 v16bf;
typedef __attribute__((ext_vector_type(8)))  float  v8f;

union FragBF {
    v16bf v;
    uint4 q[2];
    __bf16 e[16];
};

// ---------------------------------------------------------------------------
// CDNA5 async global->LDS copy (ASYNCcnt path), with safe fallback
// Builtin signature (probe-confirmed by diagnostic): first param is a pointer
// to a 4 x i32 vector in AS(1), second in AS(3), then imm offset, imm cpol.
// ---------------------------------------------------------------------------
#if __has_builtin(__builtin_amdgcn_global_load_async_to_lds_b128)
#define USE_ASYNC_LDS 1
typedef int v4i_async __attribute__((ext_vector_type(4)));
typedef __attribute__((address_space(1))) v4i_async as1_v4i;
typedef __attribute__((address_space(3))) v4i_async as3_v4i;
#endif

__device__ __forceinline__ void copy16(const __bf16* g, __bf16* l) {
#ifdef USE_ASYNC_LDS
    __builtin_amdgcn_global_load_async_to_lds_b128(
        (as1_v4i*)(void*)g, (as3_v4i*)(void*)l, 0, 0);
#else
    *(uint4*)l = *(const uint4*)g;
#endif
}

template <int N>
__device__ __forceinline__ void async_wait() {
#ifdef USE_ASYNC_LDS
#if __has_builtin(__builtin_amdgcn_s_wait_asynccnt)
    __builtin_amdgcn_s_wait_asynccnt((short)N);
#else
    asm volatile("s_wait_asynccnt %0" :: "i"(N) : "memory");
#endif
#endif
}

// ---------------------------------------------------------------------------
// WMMA fragment loaders (wave32 operand layouts per CDNA5 ISA 7.12.2)
//
// A operand (and B operand when the matrix is stored (N,K) row-major):
//   lane r = lane&15 selects row (M or N), half = lane>>4 selects K phase.
//   element e (0..15):  K = k0 + half*8 + e          (e < 8)
//                       K = k0 + 16 + half*8 + (e-8) (e >= 8)
// K must be contiguous -> two 16-byte loads per lane.
// ---------------------------------------------------------------------------
__device__ __forceinline__ v16bf frag_load(const __bf16* p, int ld,
                                           int row0, int k0, int nrows, int lane) {
    int half = (lane >> 4) & 1;
    int r    = lane & 15;
    int row  = row0 + r;
    FragBF f;
    if (row < nrows) {
        const __bf16* s = p + (size_t)row * ld + k0 + half * 8;
        f.q[0] = *(const uint4*)(s);
        f.q[1] = *(const uint4*)(s + 16);
    } else {
        f.q[0] = make_uint4(0u, 0u, 0u, 0u);
        f.q[1] = make_uint4(0u, 0u, 0u, 0u);
    }
    return f.v;
}

// B operand when the matrix is stored (K,N) row-major (strided K gather).
__device__ __forceinline__ v16bf frag_load_kn(const __bf16* p, int ld,
                                              int n0, int k0, int krows, int lane) {
    int half = (lane >> 4) & 1;
    int n    = n0 + (lane & 15);
    FragBF f;
#pragma unroll
    for (int e = 0; e < 16; ++e) {
        int koff = (e < 8) ? (half * 8 + e) : (16 + half * 8 + (e - 8));
        int k = k0 + koff;
        f.e[e] = (k < krows) ? p[(size_t)k * ld + n] : (__bf16)0.0f;
    }
    return f.v;
}

__device__ __forceinline__ v8f wmma_bf16(v16bf a, v16bf b, v8f c) {
    return __builtin_amdgcn_wmma_f32_16x16x32_bf16(false, a, false, b,
                                                   (short)0, c, false, false);
}

// ---------------------------------------------------------------------------
// K0a: x (B,149,768) f32 -> xbT (B,768,160) bf16, transposed + zero K-pad
// ---------------------------------------------------------------------------
__global__ void k_cvt_xT(const float* __restrict__ x, __bf16* __restrict__ xbT) {
    size_t i = (size_t)blockIdx.x * 256 + threadIdx.x;
    size_t total = (size_t)BATCH * T_DIM * K_PAD;
    if (i >= total) return;
    int l = (int)(i % K_PAD);
    size_t rest = i / K_PAD;
    int d = (int)(rest % T_DIM);
    int b = (int)(rest / T_DIM);
    float v = (l < T_LEN) ? x[((size_t)b * T_LEN + l) * T_DIM + d] : 0.0f;
    xbT[i] = (__bf16)v;
}

// K0b: generic f32 -> bf16 with column pad: dst[r*dc+c] = c<sc ? src[r*sc+c] : 0
__global__ void k_cvt_pad(const float* __restrict__ src, __bf16* __restrict__ dst,
                          int rows, int sc, int dc) {
    size_t i = (size_t)blockIdx.x * 256 + threadIdx.x;
    size_t total = (size_t)rows * dc;
    if (i >= total) return;
    int c = (int)(i % dc);
    size_t r = i / dc;
    dst[i] = (c < sc) ? (__bf16)src[r * sc + c] : (__bf16)0.0f;
}

// ---------------------------------------------------------------------------
// K1: fused conv1-GEMM + LayerNorm + pe_wave  -> nx bf16 (B,3136,768)
//   xe[o,d] = sum_l Wc[o,l] * x[l,d]   (b_conv1 cancels under LN)
//   One block = 16 rows (o) x full 768 cols. 8 waves x 6 N-tiles, K = 160.
// ---------------------------------------------------------------------------
__global__ void __launch_bounds__(256)
k_gemm1_ln(const __bf16* __restrict__ Wc,    // (3136,160)
           const __bf16* __restrict__ xbT,   // (B,768,160)
           const float* __restrict__ g, const float* __restrict__ beta,
           const float* __restrict__ pe,     // (768,3136)
           __bf16* __restrict__ nx) {        // (B,3136,768)
    int b    = blockIdx.x / (NSP / 16);
    int row0 = (blockIdx.x % (NSP / 16)) * 16;
    int wave = threadIdx.x >> 5;
    int lane = threadIdx.x & 31;
    int half = (lane >> 4) & 1;
    int r    = lane & 15;

    const __bf16* Bm = xbT + (size_t)b * T_DIM * K_PAD;

    __shared__ __align__(16) float S[16][T_DIM];

#pragma unroll 1
    for (int t = 0; t < 6; ++t) {
        int col0 = (wave * 6 + t) * 16;
        v8f acc = {0.f, 0.f, 0.f, 0.f, 0.f, 0.f, 0.f, 0.f};
#pragma unroll
        for (int kk = 0; kk < K_PAD; kk += 32) {
            v16bf a  = frag_load(Wc, K_PAD, row0, kk, 1 << 30, lane);
            v16bf bb = frag_load(Bm, K_PAD, col0, kk, 1 << 30, lane);
            acc = wmma_bf16(a, bb, acc);
        }
#pragma unroll
        for (int v = 0; v < 8; ++v)
            S[half * 8 + v][col0 + r] = acc[v];
    }
    __syncthreads();

    // LayerNorm over 768, two rows per wave
#pragma unroll
    for (int rr = 0; rr < 2; ++rr) {
        int row = wave * 2 + rr;
        float s = 0.f, ss = 0.f;
        for (int c = lane; c < T_DIM; c += 32) {
            float v = S[row][c];
            s += v; ss += v * v;
        }
#pragma unroll
        for (int off = 16; off > 0; off >>= 1) {
            s  += __shfl_xor(s,  off, 32);
            ss += __shfl_xor(ss, off, 32);
        }
        float mean = s * (1.0f / T_DIM);
        float var  = ss * (1.0f / T_DIM) - mean * mean;
        float ri   = rsqrtf(var + LN_EPS);
        int n = row0 + row;
        for (int c = lane; c < T_DIM; c += 32) {
            float o = (S[row][c] - mean) * ri * g[c] + beta[c]
                      + pe[(size_t)c * NSP + n];
            nx[((size_t)b * NSP + n) * T_DIM + c] = (__bf16)o;
        }
    }
}

// ---------------------------------------------------------------------------
// K2: LN over channels of y (B,512,3136) + pe_spec -> ny bf16 (B,3136,512)
//   block = 256 threads, 64 spatial tokens, 4-way channel parallel
// ---------------------------------------------------------------------------
__global__ void __launch_bounds__(256)
k_ny(const float* __restrict__ y,
     const float* __restrict__ g, const float* __restrict__ beta,
     const float* __restrict__ pe,   // (512,3136)
     __bf16* __restrict__ ny) {
    int b  = blockIdx.x / (NSP / 64);
    int n0 = (blockIdx.x % (NSP / 64)) * 64;
    int nloc = threadIdx.x & 63;
    int cpar = threadIdx.x >> 6;

    const float* yb = y + (size_t)b * S_DIM * NSP;

    float s = 0.f, ss = 0.f;
    for (int c = cpar; c < S_DIM; c += 4) {
        float v = yb[(size_t)c * NSP + n0 + nloc];
        s += v; ss += v * v;
    }
    __shared__ float rs[4][64], rss[4][64];
    rs[cpar][nloc] = s;  rss[cpar][nloc] = ss;
    __syncthreads();
    float tot  = rs[0][nloc] + rs[1][nloc] + rs[2][nloc] + rs[3][nloc];
    float tots = rss[0][nloc] + rss[1][nloc] + rss[2][nloc] + rss[3][nloc];
    float mean = tot * (1.0f / S_DIM);
    float var  = tots * (1.0f / S_DIM) - mean * mean;
    float ri   = rsqrtf(var + LN_EPS);

    for (int c = cpar; c < S_DIM; c += 4) {
        float v = yb[(size_t)c * NSP + n0 + nloc];
        float o = (v - mean) * ri * g[c] + beta[c] + pe[(size_t)c * NSP + n0 + nloc];
        ny[((size_t)b * NSP + n0 + nloc) * S_DIM + c] = (__bf16)o;
    }
}

// ---------------------------------------------------------------------------
// K3: WMMA GEMM  C = A * Bm^T + bias  (bf16 in, f32 acc, bf16 out)
//   A: (M,K) row-major; Bm: (N,K) row-major; C: (M,N) bf16. M%128==0, N%64==0,
//   K%32==0.  Block computes 128x64.
//   Double-buffered LDS staging fed by GLOBAL_LOAD_ASYNC_TO_LDS_B128
//   (ASYNCcnt / s_wait_asynccnt) so global latency hides behind the WMMAs and
//   the 8 waves stop issuing redundant B-fragment loads.
// ---------------------------------------------------------------------------
__global__ void __launch_bounds__(256)
k_gemm_nt(const __bf16* __restrict__ A, const __bf16* __restrict__ Bm,
          const float* __restrict__ bias, __bf16* __restrict__ C,
          int M, int N, int K) {
    int tiles_n = N / 64;
    int bm = blockIdx.x / tiles_n;
    int bn = blockIdx.x % tiles_n;
    int tid  = threadIdx.x;
    int wave = tid >> 5;
    int lane = tid & 31;
    int half = (lane >> 4) & 1;
    int r    = lane & 15;
    int row0 = bm * 128;

    __shared__ __align__(16) __bf16 ldsA[2][128][32];  // 2 x 8 KB
    __shared__ __align__(16) __bf16 ldsB[2][64][32];   // 2 x 4 KB

    // copy assignment: thread t moves three 16B chunks per K-stage
    int crow = tid >> 2;           // 0..63
    int cofs = (tid & 3) * 8;      // bf16 offset within the 32-wide K slice
    const __bf16* gA0 = A  + (size_t)(row0 + crow)      * K + cofs;
    const __bf16* gA1 = A  + (size_t)(row0 + 64 + crow) * K + cofs;
    const __bf16* gB  = Bm + (size_t)(bn * 64 + crow)   * K + cofs;

    int nk = K / 32;
    auto issue = [&](int s) {
        int buf = s & 1;
        int kk  = s * 32;
        copy16(gA0 + kk, &ldsA[buf][crow][cofs]);
        copy16(gA1 + kk, &ldsA[buf][64 + crow][cofs]);
        copy16(gB  + kk, &ldsB[buf][crow][cofs]);
    };

    v8f acc[4];
#pragma unroll
    for (int t = 0; t < 4; ++t)
        acc[t] = (v8f){0.f, 0.f, 0.f, 0.f, 0.f, 0.f, 0.f, 0.f};

    issue(0);
#pragma unroll 1
    for (int s = 0; s < nk; ++s) {
        if (s + 1 < nk) {
            issue(s + 1);
            async_wait<3>();   // keep only the 3 just-issued copies in flight
        } else {
            async_wait<0>();
        }
        __syncthreads();       // stage s resident for all waves

        int buf = s & 1;
        v16bf a = frag_load(&ldsA[buf][0][0], 32, wave * 16, 0, 1 << 30, lane);
#pragma unroll
        for (int t = 0; t < 4; ++t) {
            v16bf bb = frag_load(&ldsB[buf][0][0], 32, t * 16, 0, 1 << 30, lane);
            acc[t] = wmma_bf16(a, bb, acc[t]);
        }
        __syncthreads();       // reads done before buf is overwritten again
    }

#pragma unroll
    for (int t = 0; t < 4; ++t) {
        int col = bn * 64 + t * 16 + r;
        float bs = bias ? bias[col] : 0.f;
#pragma unroll
        for (int v = 0; v < 8; ++v) {
            int row = row0 + wave * 16 + half * 8 + v;
            C[(size_t)row * N + col] = (__bf16)(acc[t][v] + bs);
        }
    }
}

// ---------------------------------------------------------------------------
// K4: row-local attention, one block per (b,h) row of 56 tokens (padded to 64)
//   S = q k^T / sqrt(512)  (WMMA, K=512)
//   softmax (f32, padded keys masked)  -> P bf16 in LDS
//   O = P @ V (WMMA, K=64) ; epilogue: NHWC->NCHW + residual y  -> d_out
// ---------------------------------------------------------------------------
__global__ void __launch_bounds__(256)
k_attn(const __bf16* __restrict__ q, const __bf16* __restrict__ k,
       const __bf16* __restrict__ v, const float* __restrict__ y,
       float* __restrict__ out) {
    int b = blockIdx.x / HW;
    int h = blockIdx.x % HW;
    int wave = threadIdx.x >> 5;
    int lane = threadIdx.x & 31;
    int half = (lane >> 4) & 1;
    int r    = lane & 15;

    const __bf16* qr = q + ((size_t)b * NSP + h * HW) * S_DIM;
    const __bf16* kr = k + ((size_t)b * NSP + h * HW) * S_DIM;
    const __bf16* vr = v + ((size_t)b * NSP + h * HW) * S_DIM;

    __shared__ __align__(16) float  S[64][64];
    __shared__ __align__(16) __bf16 P[64][64];

    // ---- phase 1: scores (4x4 tiles of 16x16, 2 tiles per wave) ----
    const float scale = 0.04419417382415922f;  // 1/sqrt(512)
#pragma unroll
    for (int t = 0; t < 2; ++t) {
        int tile = wave * 2 + t;
        int mi = tile >> 2, ni = tile & 3;
        v8f acc = {0.f, 0.f, 0.f, 0.f, 0.f, 0.f, 0.f, 0.f};
        for (int kk = 0; kk < S_DIM; kk += 32) {
            v16bf a  = frag_load(qr, S_DIM, mi * 16, kk, HW, lane);
            v16bf bb = frag_load(kr, S_DIM, ni * 16, kk, HW, lane);
            acc = wmma_bf16(a, bb, acc);
        }
#pragma unroll
        for (int e = 0; e < 8; ++e)
            S[mi * 16 + half * 8 + e][ni * 16 + r] = acc[e] * scale;
    }
    __syncthreads();

    // ---- softmax: one thread per query row; mask padded keys ----
    if (threadIdx.x < 64) {
        int row = threadIdx.x;
        float m = -1e30f;
        for (int c = 0; c < HW; ++c) m = fmaxf(m, S[row][c]);
        float sum = 0.f;
        for (int c = 0; c < HW; ++c) {
            float e = __expf(S[row][c] - m);
            S[row][c] = e;
            sum += e;
        }
        float inv = 1.0f / sum;
        for (int c = 0; c < 64; ++c)
            P[row][c] = (c < HW) ? (__bf16)(S[row][c] * inv) : (__bf16)0.0f;
    }
    __syncthreads();

    // ---- phase 2: O = P @ V, fused residual + NCHW store ----
#pragma unroll 1
    for (int mi = 0; mi < 4; ++mi) {
#pragma unroll 1
        for (int nn = 0; nn < 4; ++nn) {
            int ni = wave * 4 + nn;
            int col0 = ni * 16;
            v8f acc = {0.f, 0.f, 0.f, 0.f, 0.f, 0.f, 0.f, 0.f};
#pragma unroll
            for (int kk = 0; kk < 64; kk += 32) {
                v16bf a  = frag_load(&P[0][0], 64, mi * 16, kk, 64, lane);
                v16bf bb = frag_load_kn(vr, S_DIM, col0, kk, HW, lane);
                acc = wmma_bf16(a, bb, acc);
            }
#pragma unroll
            for (int e = 0; e < 8; ++e) {
                int wrow = mi * 16 + half * 8 + e;
                if (wrow < HW) {
                    size_t o = (((size_t)b * S_DIM + col0 + r) * HW + h) * HW + wrow;
                    out[o] = acc[e] + y[o];
                }
            }
        }
    }
}

// ---------------------------------------------------------------------------
// Host launcher
// ---------------------------------------------------------------------------
extern "C" void kernel_launch(void* const* d_in, const int* in_sizes, int n_in,
                              void* d_out, int out_size, void* d_ws, size_t ws_size,
                              hipStream_t stream) {
    const float* x    = (const float*)d_in[0];
    const float* y    = (const float*)d_in[1];
    const float* Wc1  = (const float*)d_in[2];
    // d_in[3] = b_conv1 : constant per output row -> cancels exactly under LN
    const float* ln1g = (const float*)d_in[4];
    const float* ln1b = (const float*)d_in[5];
    const float* ln2g = (const float*)d_in[6];
    const float* ln2b = (const float*)d_in[7];
    const float* pew  = (const float*)d_in[8];
    const float* pes  = (const float*)d_in[9];
    const float* Wq   = (const float*)d_in[10];
    const float* bq   = (const float*)d_in[11];
    const float* Wk   = (const float*)d_in[12];
    const float* bk   = (const float*)d_in[13];
    float* out = (float*)d_out;

    char* wsp = (char*)d_ws;
    size_t off = 0;
    auto carve = [&](size_t bytes) -> void* {
        void* p = wsp + off;
        off += (bytes + 255) & ~(size_t)255;
        return p;
    };
    __bf16* xbT = (__bf16*)carve((size_t)BATCH * T_DIM * K_PAD * 2);
    __bf16* Wcb = (__bf16*)carve((size_t)NSP * K_PAD * 2);
    __bf16* Wqb = (__bf16*)carve((size_t)S_DIM * T_DIM * 2);
    __bf16* Wkb = (__bf16*)carve((size_t)S_DIM * S_DIM * 2);
    __bf16* nx  = (__bf16*)carve((size_t)BATCH * NSP * T_DIM * 2);
    __bf16* ny  = (__bf16*)carve((size_t)BATCH * NSP * S_DIM * 2);
    __bf16* qb  = (__bf16*)carve((size_t)BATCH * NSP * S_DIM * 2);
    __bf16* kb  = (__bf16*)carve((size_t)BATCH * NSP * S_DIM * 2);

    // K0: precision conversion / padding / transpose
    {
        size_t n = (size_t)BATCH * T_DIM * K_PAD;
        k_cvt_xT<<<(unsigned)((n + 255) / 256), 256, 0, stream>>>(x, xbT);
    }
    {
        size_t n = (size_t)NSP * K_PAD;
        k_cvt_pad<<<(unsigned)((n + 255) / 256), 256, 0, stream>>>(Wc1, Wcb, NSP, T_LEN, K_PAD);
    }
    {
        size_t n = (size_t)S_DIM * T_DIM;
        k_cvt_pad<<<(unsigned)((n + 255) / 256), 256, 0, stream>>>(Wq, Wqb, S_DIM, T_DIM, T_DIM);
    }
    {
        size_t n = (size_t)S_DIM * S_DIM;
        k_cvt_pad<<<(unsigned)((n + 255) / 256), 256, 0, stream>>>(Wk, Wkb, S_DIM, S_DIM, S_DIM);
    }

    // K1: conv1 GEMM + LN + pe_wave -> nx
    k_gemm1_ln<<<BATCH * (NSP / 16), 256, 0, stream>>>(Wcb, xbT, ln1g, ln1b, pew, nx);

    // K2: LN(y) + pe_spec -> ny (channel-last bf16; also serves as V)
    k_ny<<<BATCH * (NSP / 64), 256, 0, stream>>>(y, ln2g, ln2b, pes, ny);

    // K3: q = nx @ Wq^T + bq ; k = ny @ Wk^T + bk  (async-LDS WMMA GEMM)
    {
        int M = BATCH * NSP;
        k_gemm_nt<<<(M / 128) * (S_DIM / 64), 256, 0, stream>>>(nx, Wqb, bq, qb, M, S_DIM, T_DIM);
        k_gemm_nt<<<(M / 128) * (S_DIM / 64), 256, 0, stream>>>(ny, Wkb, bk, kb, M, S_DIM, S_DIM);
    }

    // K4: row-local attention + residual -> out (B,512,56,56)
    k_attn<<<BATCH * HW, 256, 0, stream>>>(qb, kb, ny, y, out);

    (void)in_sizes; (void)n_in; (void)out_size; (void)ws_size;
}
